// CasualAttention_53386443489951
// MI455X (gfx1250) — matrix operand order repaired and checked by reading, not compiled, not run on hardware
//
#include <hip/hip_runtime.h>

typedef __bf16 bf16;
typedef __attribute__((ext_vector_type(16))) __bf16 v16bf;
typedef __attribute__((ext_vector_type(8)))  __bf16 v8bf;
typedef __attribute__((ext_vector_type(4)))  __bf16 v4bf;
typedef __attribute__((ext_vector_type(8)))  float   v8f;
typedef __attribute__((ext_vector_type(4)))  float   v4f;

#define WMMA_BF16(Am, Bm, Cm) \
  __builtin_amdgcn_wmma_f32_16x16x32_bf16(false, (Am), false, (Bm), (short)0, (Cm), false, false)

// CDNA5 async copy global -> LDS (ASYNCcnt-tracked), per-lane 16B.
#define ASYNC_B128(ldsaddr, gptr) \
  asm volatile("global_load_async_to_lds_b128 %0, %1, off" :: "v"(ldsaddr), "v"(gptr) : "memory")
#define WAIT_ASYNC() asm volatile("s_wait_asynccnt 0x0" ::: "memory")

// Generic pointers to LDS carry the LDS byte address in their low 32 bits.
__device__ __forceinline__ unsigned lds_off(const void* p) {
  return (unsigned)(unsigned long long)p;
}

union BFrag { v16bf v; v8bf h[2]; };

constexpr int Bb = 2, Ss = 2048, Dd = 2048, Hh = 16, HDd = 128;
constexpr int BM = 128, BN = 128, BK = 64, LDA = BK + 8;

// ---------------------------------------------------------------------------
// f32 -> bf16 pre-convert (grid-stride, float4 granularity)
// ---------------------------------------------------------------------------
__global__ __launch_bounds__(256)
void cvt_bf16(const float* __restrict__ in, bf16* __restrict__ out, int n4)
{
  const int stride = gridDim.x * 256;
  for (int i = blockIdx.x * 256 + threadIdx.x; i < n4; i += stride) {
    v4f x = *(const v4f*)(in + (size_t)i * 4);
    v4bf o; o[0]=(bf16)x[0]; o[1]=(bf16)x[1]; o[2]=(bf16)x[2]; o[3]=(bf16)x[3];
    *(v4bf*)(out + (size_t)i * 4) = o;
  }
}

// ---------------------------------------------------------------------------
// GEMM: C = X @ W^T + bias.  X,W bf16 row-major (k-contiguous both sides).
// Double-buffered async-LDS staging.  128x128x64 tile, 8 waves, 32x64/wave.
// mode 0: RoPE epilogue, bf16 row-major (Q/K)
// mode 1: bf16 transposed per (b,h): [b][h][hd][s]  (V^T)
// mode 2: f32 row-major (final projection)
// ---------------------------------------------------------------------------
__global__ __launch_bounds__(256)
void gemm_xwt(const bf16* __restrict__ X, const bf16* __restrict__ W,
              const float* __restrict__ bias, const float* __restrict__ fc,
              const float* __restrict__ fs, bf16* __restrict__ outB,
              float* __restrict__ outF, int mode)
{
  __shared__ bf16 As[2][BM * LDA];
  __shared__ bf16 Bs[2][BN * LDA];
  const int tid   = threadIdx.x;
  const int lane  = tid & 31, wid = tid >> 5;
  const int lrow  = lane & 15, lhalf = lane >> 4;
  const int wm0   = (wid >> 1) * 32, wn0 = (wid & 1) * 64;
  const int m0    = blockIdx.y * BM, n0 = blockIdx.x * BN;

  auto load_tiles = [&](int k0, int buf) {
    #pragma unroll
    for (int i = 0; i < 4; ++i) {                 // A: 128 rows x 8 chunks
      int j = tid + 256 * i;
      int r = j >> 3, c = (j & 7) * 8;
      ASYNC_B128(lds_off(&As[buf][r * LDA + c]),
                 X + (size_t)(m0 + r) * Dd + k0 + c);
    }
    #pragma unroll
    for (int i = 0; i < 4; ++i) {                 // B: 128 rows x 8 chunks
      int j = tid + 256 * i;
      int r = j >> 3, c = (j & 7) * 8;
      ASYNC_B128(lds_off(&Bs[buf][r * LDA + c]),
                 W + (size_t)(n0 + r) * Dd + k0 + c);
    }
  };

  v8f acc[2][4] = {};

  load_tiles(0, 0);
  WAIT_ASYNC();
  __syncthreads();

  int buf = 0;
  for (int k0 = 0; k0 < Dd; k0 += BK, buf ^= 1) {
    if (k0 + BK < Dd) load_tiles(k0 + BK, buf ^ 1);   // prefetch next tile

    #pragma unroll
    for (int kk = 0; kk < BK; kk += 32) {
      BFrag af[2], bfr[4];
      #pragma unroll
      for (int mt = 0; mt < 2; ++mt) {
        const bf16* p = &As[buf][(wm0 + mt * 16 + lrow) * LDA + kk + lhalf * 8];
        af[mt].h[0] = *(const v8bf*)p;
        af[mt].h[1] = *(const v8bf*)(p + 16);
      }
      #pragma unroll
      for (int nt = 0; nt < 4; ++nt) {
        const bf16* p = &Bs[buf][(wn0 + nt * 16 + lrow) * LDA + kk + lhalf * 8];
        bfr[nt].h[0] = *(const v8bf*)p;
        bfr[nt].h[1] = *(const v8bf*)(p + 16);
      }
      #pragma unroll
      for (int mt = 0; mt < 2; ++mt)
        #pragma unroll
        for (int nt = 0; nt < 4; ++nt)
          acc[mt][nt] = WMMA_BF16(af[mt].v, bfr[nt].v, acc[mt][nt]);
    }
    WAIT_ASYNC();
    __syncthreads();
  }

  // epilogue
  #pragma unroll
  for (int mt = 0; mt < 2; ++mt) {
    #pragma unroll
    for (int nt = 0; nt < 4; ++nt) {
      const int col  = n0 + wn0 + nt * 16 + lrow;
      const float bc = bias[col];
      v8f c = acc[mt][nt];
      #pragma unroll
      for (int vg = 0; vg < 8; ++vg) {
        const int row = m0 + wm0 + mt * 16 + vg + lhalf * 8;
        float val = c[vg] + bc;
        if (mode == 0) {                       // RoPE, bf16 row-major
          const float other = __shfl_xor(val, 1);
          const int i = (col & 127) >> 1;
          const int s = row & (Ss - 1);
          const float cv = fc[s * 64 + i], sv = fs[s * 64 + i];
          val = (col & 1) ? (other * sv + val * cv) : (val * cv - other * sv);
          outB[(size_t)row * Dd + col] = (bf16)val;
        } else if (mode == 1) {                // V^T: [b][h][hd][s]
          const int bb = row >> 11, s = row & 2047;
          const int hh = col >> 7, hd = col & 127;
          outB[(((size_t)(bb * Hh + hh) * HDd + hd) << 11) + s] = (bf16)val;
        } else {                               // f32 row-major
          outF[(size_t)row * Dd + col] = val;
        }
      }
    }
  }
}

// ---------------------------------------------------------------------------
// Flash attention, causal.  Grid: (S/64, B*H).  4 waves, each owns 16 q rows.
// K/V tiles staged via async-to-LDS, double buffered, shared by all waves.
// Block-uniform K loop (masked tiles contribute exactly zero), so barriers
// stay convergent.  Online softmax in exp2 with 1/sqrt(128) folded in.
// ---------------------------------------------------------------------------
__global__ __launch_bounds__(128)
void flash_attn(const bf16* __restrict__ Q, const bf16* __restrict__ K,
                const bf16* __restrict__ Vt, bf16* __restrict__ O)
{
  constexpr int LDK = 136, LDV = 72, LDP = 72;
  __shared__ bf16 Kt[2][64 * LDK];    // 64 keys x 128 hd
  __shared__ bf16 Vb[2][128 * LDV];   // 128 hd x 64 keys (from V^T)
  __shared__ bf16 Pbuf[4][16 * LDP];  // per-wave P scratch
  const int tid  = threadIdx.x;
  const int wid  = tid >> 5, lane = tid & 31;
  const int lrow = lane & 15, lhalf = lane >> 4;
  const int b = blockIdx.y >> 4, h = blockIdx.y & 15;
  const int qt0 = blockIdx.x * 64;
  const int q0  = qt0 + wid * 16;
  const float kscale = 1.44269504089f * 0.08838834764832f;  // log2(e)/sqrt(128)

  auto load_kv = [&](int k0, int buf) {
    const bf16* kbase = K + ((size_t)(b * Ss + k0)) * Dd + h * HDd;
    #pragma unroll
    for (int i = 0; i < 8; ++i) {                 // 64 rows x 16 chunks
      int j = tid + 128 * i;
      int r = j >> 4, c = (j & 15) * 8;
      ASYNC_B128(lds_off(&Kt[buf][r * LDK + c]), kbase + (size_t)r * Dd + c);
    }
    const bf16* vbase = Vt + ((size_t)(b * Hh + h) * HDd) * Ss + k0;
    #pragma unroll
    for (int i = 0; i < 8; ++i) {                 // 128 rows x 8 chunks
      int j = tid + 128 * i;
      int r = j >> 3, c = (j & 7) * 8;
      ASYNC_B128(lds_off(&Vb[buf][r * LDV + c]), vbase + (size_t)r * Ss + c);
    }
  };

  load_kv(0, 0);

  // Q fragments for this wave's 16 rows (loaded once, reused)
  BFrag qf[4];
  {
    const bf16* qrow = Q + ((size_t)(b * Ss + q0 + lrow)) * Dd + h * HDd;
    #pragma unroll
    for (int kc = 0; kc < 4; ++kc) {
      const bf16* p = qrow + kc * 32 + lhalf * 8;
      qf[kc].h[0] = *(const v8bf*)p;
      qf[kc].h[1] = *(const v8bf*)(p + 16);
    }
  }

  v8f oacc[8] = {};
  float mrun[8], lrun[8];
  #pragma unroll
  for (int vg = 0; vg < 8; ++vg) { mrun[vg] = -1e30f; lrun[vg] = 0.f; }
  bf16* P = Pbuf[wid];

  WAIT_ASYNC();
  __syncthreads();

  const int kend = qt0 + 64;
  int buf = 0;
  for (int k0 = 0; k0 < kend; k0 += 64, buf ^= 1) {
    if (k0 + 64 < kend) load_kv(k0 + 64, buf ^ 1);    // prefetch next K/V tile

    // --- S = Q @ K^T (16 x 64), 16 WMMAs from LDS ---
    v8f sacc[4] = {};
    #pragma unroll
    for (int nt = 0; nt < 4; ++nt) {
      #pragma unroll
      for (int kc = 0; kc < 4; ++kc) {
        BFrag kf;
        const bf16* p = &Kt[buf][(nt * 16 + lrow) * LDK + kc * 32 + lhalf * 8];
        kf.h[0] = *(const v8bf*)p;
        kf.h[1] = *(const v8bf*)(p + 16);
        sacc[nt] = WMMA_BF16(qf[kc].v, kf.v, sacc[nt]);
      }
    }
    // --- causal mask ---
    #pragma unroll
    for (int nt = 0; nt < 4; ++nt) {
      const int col = k0 + nt * 16 + lrow;
      #pragma unroll
      for (int vg = 0; vg < 8; ++vg) {
        const int row = q0 + vg + lhalf * 8;
        sacc[nt][vg] = (col > row) ? -1e30f : sacc[nt][vg];
      }
    }
    // --- online softmax state update ---
    #pragma unroll
    for (int vg = 0; vg < 8; ++vg) {
      float t = fmaxf(fmaxf(sacc[0][vg], sacc[1][vg]), fmaxf(sacc[2][vg], sacc[3][vg]));
      t = fmaxf(t, __shfl_xor(t, 1));
      t = fmaxf(t, __shfl_xor(t, 2));
      t = fmaxf(t, __shfl_xor(t, 4));
      t = fmaxf(t, __shfl_xor(t, 8));
      const float mnew = fmaxf(mrun[vg], t);
      const float f = __builtin_exp2f((mrun[vg] - mnew) * kscale);
      mrun[vg] = mnew;
      lrun[vg] *= f;
      #pragma unroll
      for (int dt = 0; dt < 8; ++dt) oacc[dt][vg] *= f;
    }
    // --- probabilities + row sums ---
    #pragma unroll
    for (int nt = 0; nt < 4; ++nt)
      #pragma unroll
      for (int vg = 0; vg < 8; ++vg)
        sacc[nt][vg] = __builtin_exp2f((sacc[nt][vg] - mrun[vg]) * kscale);
    #pragma unroll
    for (int vg = 0; vg < 8; ++vg) {
      float rs = (sacc[0][vg] + sacc[1][vg]) + (sacc[2][vg] + sacc[3][vg]);
      rs += __shfl_xor(rs, 1);
      rs += __shfl_xor(rs, 2);
      rs += __shfl_xor(rs, 4);
      rs += __shfl_xor(rs, 8);
      lrun[vg] += rs;
    }
    // --- P -> per-wave LDS (C layout -> A layout), same-wave DS is in-order ---
    #pragma unroll
    for (int nt = 0; nt < 4; ++nt)
      #pragma unroll
      for (int vg = 0; vg < 8; ++vg)
        P[(vg + lhalf * 8) * LDP + nt * 16 + lrow] = (bf16)sacc[nt][vg];

    BFrag pf[2];
    #pragma unroll
    for (int kc = 0; kc < 2; ++kc) {
      const bf16* p = P + lrow * LDP + kc * 32 + lhalf * 8;
      pf[kc].h[0] = *(const v8bf*)p;
      pf[kc].h[1] = *(const v8bf*)(p + 16);
    }
    // --- O += P @ V (16 x 128), 16 WMMAs from LDS ---
    #pragma unroll
    for (int dt = 0; dt < 8; ++dt) {
      #pragma unroll
      for (int kc = 0; kc < 2; ++kc) {
        BFrag vf;
        const bf16* p = &Vb[buf][(dt * 16 + lrow) * LDV + kc * 32 + lhalf * 8];
        vf.h[0] = *(const v8bf*)p;
        vf.h[1] = *(const v8bf*)(p + 16);
        oacc[dt] = WMMA_BF16(pf[kc].v, vf.v, oacc[dt]);
      }
    }
    WAIT_ASYNC();
    __syncthreads();
  }

  // normalize + store bf16 [b][s][h*128+hd]
  #pragma unroll
  for (int vg = 0; vg < 8; ++vg) {
    const float inv = 1.0f / lrun[vg];
    const int row = q0 + vg + lhalf * 8;
    bf16* orow = O + ((size_t)(b * Ss + row)) * Dd + h * HDd;
    #pragma unroll
    for (int dt = 0; dt < 8; ++dt)
      orow[dt * 16 + lrow] = (bf16)(oacc[dt][vg] * inv);
  }
}

// ---------------------------------------------------------------------------
extern "C" void kernel_launch(void* const* d_in, const int* in_sizes, int n_in,
                              void* d_out, int out_size, void* d_ws, size_t ws_size,
                              hipStream_t stream)
{
  const float* x  = (const float*)d_in[0];
  const float* wq = (const float*)d_in[1];
  const float* bq = (const float*)d_in[2];
  const float* wk = (const float*)d_in[3];
  const float* bk = (const float*)d_in[4];
  const float* wv = (const float*)d_in[5];
  const float* bv = (const float*)d_in[6];
  const float* wp = (const float*)d_in[7];
  const float* bp = (const float*)d_in[8];
  const float* fc = (const float*)d_in[9];
  const float* fs = (const float*)d_in[10];

  const size_t NE = (size_t)Bb * Ss * Dd;   // 8M
  const size_t NW = (size_t)Dd * Dd;        // 4M
  bf16* qws = (bf16*)d_ws;
  bf16* kws = qws + NE;
  bf16* vtw = kws + NE;
  bf16* aws = vtw + NE;
  bf16* xb  = aws + NE;
  bf16* wqb = xb  + NE;
  bf16* wkb = wqb + NW;
  bf16* wvb = wkb + NW;
  bf16* wpb = wvb + NW;

  // 1) pre-convert activations + weights to bf16 (bandwidth-trivial)
  cvt_bf16<<<2048, 256, 0, stream>>>(x,  xb,  (int)(NE / 4));
  cvt_bf16<<<2048, 256, 0, stream>>>(wq, wqb, (int)(NW / 4));
  cvt_bf16<<<2048, 256, 0, stream>>>(wk, wkb, (int)(NW / 4));
  cvt_bf16<<<2048, 256, 0, stream>>>(wv, wvb, (int)(NW / 4));
  cvt_bf16<<<2048, 256, 0, stream>>>(wp, wpb, (int)(NW / 4));

  dim3 gg(Dd / BN, (Bb * Ss) / BM);         // (16, 32)
  // 2) QKV projections (+RoPE, +V^T layout)
  gemm_xwt<<<gg, 256, 0, stream>>>(xb, wqb, bq, fc, fs, qws, nullptr, 0);
  gemm_xwt<<<gg, 256, 0, stream>>>(xb, wkb, bk, fc, fs, kws, nullptr, 0);
  gemm_xwt<<<gg, 256, 0, stream>>>(xb, wvb, bv, fc, fs, vtw, nullptr, 1);
  // 3) causal flash attention
  flash_attn<<<dim3(Ss / 64, Bb * Hh), 128, 0, stream>>>(qws, kws, vtw, aws);
  // 4) output projection -> f32
  gemm_xwt<<<gg, 256, 0, stream>>>(aws, wpb, bp, nullptr, nullptr, nullptr,
                                   (float*)d_out, 2);
}
